// MultiHeadAttention_48541720379407
// MI455X (gfx1250) — compile-verified, matrix operand order checked by reading, not run on hardware
//
#include <hip/hip_runtime.h>
#include <hip/hip_bf16.h>

// ---------------- problem constants ----------------
static constexpr int Bb   = 2;
static constexpr int Ss   = 2048;
static constexpr int Dd   = 1024;
static constexpr int Hh   = 16;
static constexpr int HD   = 64;           // head dim
static constexpr int Mtot = Bb * Ss;      // 4096 rows
static constexpr int NQKV = 3 * Dd;       // 3072

// sched_group_barrier masks
#define SGB_MFMA      0x008   // matches MFMA/WMMA
#define SGB_VMEM_READ 0x020

// ---------------- WMMA types ----------------
typedef __attribute__((ext_vector_type(16))) __bf16 v16bf;
typedef __attribute__((ext_vector_type(8)))  __bf16 v8bf;
typedef __attribute__((ext_vector_type(4)))  __bf16 v4bf;
typedef __attribute__((ext_vector_type(8)))  float  v8f;

#define WMMA_BF16(a, b, c) \
  __builtin_amdgcn_wmma_f32_16x16x32_bf16(false, (a), false, (b), (short)0, (c), false, false)

// Build a v16bf A/B operand from two contiguous 8-half (16B) runs.
// Layout (ISA 7.12.2, 16-bit 16x32): lane half g needs K = [g*8, g*8+8) in
// elements 0..7 and K = [16+g*8, 16+g*8+8) in elements 8..15.
static __device__ __forceinline__ v16bf mk16(v8bf lo, v8bf hi) {
  v16bf r;
#pragma unroll
  for (int i = 0; i < 8; ++i) { r[i] = lo[i]; r[i + 8] = hi[i]; }
  return r;
}
static __device__ __forceinline__ v16bf lda(const __bf16* row, int g) {
  return mk16(*(const v8bf*)(row + g * 8), *(const v8bf*)(row + 16 + g * 8));
}

// ---------------- kernel 1: fp32 -> bf16 convert (float4 vectorized) --------
__global__ void k_cvt(const float4* __restrict__ src, v4bf* __restrict__ dst, int n4) {
  int i = blockIdx.x * blockDim.x + threadIdx.x;
  int stride = gridDim.x * blockDim.x;
  for (; i < n4; i += stride) {
    float4 x = src[i];
    v4bf y;
    y[0] = (__bf16)x.x; y[1] = (__bf16)x.y; y[2] = (__bf16)x.z; y[3] = (__bf16)x.w;
    dst[i] = y;
  }
}

// -------- kernel 2: LDS-tiled transpose + convert (W[k][n] -> Wt[n][k]) -----
// rows, cols multiples of 32; block = 256 threads (32 x 8)
__global__ __launch_bounds__(256) void k_tr_cvt(const float* __restrict__ src,
                                                __bf16* __restrict__ dst,
                                                int rows, int cols) {
  __shared__ float tile[32][33];
  const int tx = threadIdx.x & 31;
  const int ty = threadIdx.x >> 5;
  const int c0 = blockIdx.x * 32;
  const int r0 = blockIdx.y * 32;
#pragma unroll
  for (int i = ty; i < 32; i += 8)
    tile[i][tx] = src[(size_t)(r0 + i) * cols + (c0 + tx)];
  __syncthreads();
#pragma unroll
  for (int i = ty; i < 32; i += 8)
    dst[(size_t)(c0 + i) * rows + (r0 + tx)] = (__bf16)tile[tx][i];
}

// -------- kernel 2b: per-head bf16 transpose V[s][hd] -> V^T[hd][s] ---------
__global__ __launch_bounds__(256) void k_tr_v(const __bf16* __restrict__ src,
                                              __bf16* __restrict__ dst) {
  __shared__ __bf16 tile[32][33];
  const int tx = threadIdx.x & 31;
  const int ty = threadIdx.x >> 5;
  const __bf16* s = src + (size_t)blockIdx.z * Ss * HD;  // [S][HD]
  __bf16*       d = dst + (size_t)blockIdx.z * HD * Ss;  // [HD][S]
  const int c0 = blockIdx.x * 32;  // hd
  const int r0 = blockIdx.y * 32;  // s
#pragma unroll
  for (int i = ty; i < 32; i += 8)
    tile[i][tx] = s[(size_t)(r0 + i) * HD + (c0 + tx)];
  __syncthreads();
#pragma unroll
  for (int i = ty; i < 32; i += 8)
    d[(size_t)(c0 + i) * Ss + (r0 + tx)] = tile[tx][i];
}

// -------- shared wave-level GEMM core: 16x64 C tile, K step 32 --------
// Fully unrolled (KD compile-time): straight-line SSA -> accumulators
// coalesce in place. sched_group_barrier pipeline keeps one K-step of
// VMEM loads in flight behind every WMMA quad.
template <int KD>
static __device__ __forceinline__ void wave_gemm_16x64(
    const __bf16* __restrict__ A,   // [Mtot][KD] row major (bf16)
    const __bf16* __restrict__ Bt,  // [Ntot][KD] row major (B transposed)
    int m0, int n0, int l16, int g,
    v8f& acc0, v8f& acc1, v8f& acc2, v8f& acc3) {
  const __bf16* ar  = A + (size_t)(m0 + l16) * KD;
  const __bf16* br0 = Bt + (size_t)(n0 + l16) * KD;
  const __bf16* br1 = br0 + (size_t)16 * KD;
  const __bf16* br2 = br0 + (size_t)32 * KD;
  const __bf16* br3 = br0 + (size_t)48 * KD;

  // stage-A operands: K-step 0
  v16bf a  = lda(ar, g);
  v16bf b0 = lda(br0, g), b1 = lda(br1, g);
  v16bf b2 = lda(br2, g), b3 = lda(br3, g);

#pragma unroll
  for (int kk = 32; kk + 32 < KD; kk += 64) {
    // stage-B loads (step kk)
    v16bf c  = lda(ar + kk, g);
    v16bf d0 = lda(br0 + kk, g), d1 = lda(br1 + kk, g);
    v16bf d2 = lda(br2 + kk, g), d3 = lda(br3 + kk, g);
    acc0 = WMMA_BF16(a, b0, acc0);
    acc1 = WMMA_BF16(a, b1, acc1);
    acc2 = WMMA_BF16(a, b2, acc2);
    acc3 = WMMA_BF16(a, b3, acc3);

    // stage-A loads (step kk+32)
    a  = lda(ar + kk + 32, g);
    b0 = lda(br0 + kk + 32, g); b1 = lda(br1 + kk + 32, g);
    b2 = lda(br2 + kk + 32, g); b3 = lda(br3 + kk + 32, g);
    acc0 = WMMA_BF16(c, d0, acc0);
    acc1 = WMMA_BF16(c, d1, acc1);
    acc2 = WMMA_BF16(c, d2, acc2);
    acc3 = WMMA_BF16(c, d3, acc3);
  }
  // final stage (step KD-64 when KD/32 is even; KD-32 is never loaded)
  acc0 = WMMA_BF16(a, b0, acc0);
  acc1 = WMMA_BF16(a, b1, acc1);
  acc2 = WMMA_BF16(a, b2, acc2);
  acc3 = WMMA_BF16(a, b3, acc3);

  // ---- scheduling pipeline: [L0][L1] [W0][L2] [W1][L3] ... [W31] ----
  // K-steps: KD/32; per step: 10 VMEM b128 reads, 4 WMMAs.
  constexpr int NSTEP = KD / 32;
  __builtin_amdgcn_sched_group_barrier(SGB_VMEM_READ, 10, 0);
  __builtin_amdgcn_sched_group_barrier(SGB_VMEM_READ, 10, 0);
#pragma unroll
  for (int i = 0; i < NSTEP - 1; ++i) {
    __builtin_amdgcn_sched_group_barrier(SGB_MFMA, 4, 0);
    __builtin_amdgcn_sched_group_barrier(SGB_VMEM_READ, 10, 0);
  }
  __builtin_amdgcn_sched_group_barrier(SGB_MFMA, 4, 0);
}

// ---------------- kernel 3: fused QKV projection ----------------
// C = Xb @ Wqkv + b ; epilogue: Q scaled 1/32, K and V row-major [s][hd]
__global__ __launch_bounds__(128) void k_qkv_gemm(
    const __bf16* __restrict__ Xb, const __bf16* __restrict__ Wt,
    const float* __restrict__ bias,
    __bf16* __restrict__ Qb, __bf16* __restrict__ Kb, __bf16* __restrict__ Vrow) {
  const int lane = threadIdx.x & 31;
  const int wave = threadIdx.x >> 5;
  const int l16  = lane & 15;
  const int g    = lane >> 4;
  const int m0   = blockIdx.x * 64 + wave * 16;
  const int n0   = blockIdx.y * 64;

  v8f acc0 = {}, acc1 = {}, acc2 = {}, acc3 = {};
  wave_gemm_16x64<Dd>(Xb, Wt, m0, n0, l16, g, acc0, acc1, acc2, acc3);
  const v8f accs[4] = {acc0, acc1, acc2, acc3};

#pragma unroll
  for (int j = 0; j < 4; ++j) {
    const int ng  = n0 + j * 16 + l16;
    const float bv = bias[ng];
    const int sec = ng >> 10;        // 0:Q 1:K 2:V
    const int rem = ng & 1023;
    const int h   = rem >> 6;
    const int dd  = rem & 63;
#pragma unroll
    for (int r = 0; r < 8; ++r) {
      const int mg = m0 + 8 * g + r;
      const int b  = mg >> 11;       // / Ss
      const int s  = mg & 2047;
      const float v = accs[j][r] + bv;
      const size_t idx = (((size_t)b * Hh + h) * Ss + s) * HD + dd;
      if (sec == 0)      Qb[idx]   = (__bf16)(v * 0.03125f); // 1/sqrt(D)
      else if (sec == 1) Kb[idx]   = (__bf16)v;
      else               Vrow[idx] = (__bf16)v;              // coalesced; transposed later
    }
  }
}

// ---------------- kernel 4: flash attention (one wave per 16-query block) ---
__global__ __launch_bounds__(128) void k_attn(
    const __bf16* __restrict__ Qb, const __bf16* __restrict__ Kb,
    const __bf16* __restrict__ Vt, __bf16* __restrict__ Ab) {
  __shared__ __attribute__((aligned(16))) __bf16 plds[4][16][40]; // padded rows

  const int lane = threadIdx.x & 31;
  const int wave = threadIdx.x >> 5;
  const int l16  = lane & 15;
  const int g    = lane >> 4;

  const int task    = blockIdx.x * 4 + wave;
  const int qblocks = Ss / 16;                 // 128
  const int bh      = task / qblocks;
  const int q0      = (task % qblocks) * 16;
  const int b       = bh / Hh;
  const int h       = bh % Hh;

  const __bf16* Qh = Qb + (size_t)bh * Ss * HD;
  const __bf16* Kh = Kb + (size_t)bh * Ss * HD;
  const __bf16* Vh = Vt + (size_t)bh * HD * Ss;

  // Q block as two A operands (head-dim halves 0:31, 32:63)
  const __bf16* qrow = Qh + (size_t)(q0 + l16) * HD;
  v16bf aQ0 = lda(qrow, g);
  v16bf aQ1 = lda(qrow + 32, g);

  float mrow[8], lrow[8];
#pragma unroll
  for (int r = 0; r < 8; ++r) { mrow[r] = -__builtin_inff(); lrow[r] = 0.0f; }
  v8f o0 = {}, o1 = {}, o2 = {}, o3 = {};

  const int kv_end = q0 + 16;

  // prologue: K-tile operands for kv0 = 0
  const __bf16* kr0 = Kh + (size_t)l16 * HD;
  const __bf16* kr1 = Kh + (size_t)(16 + l16) * HD;
  v16bf bK0 = lda(kr0, g),      bK1 = lda(kr0 + 32, g);
  v16bf bK2 = lda(kr1, g),      bK3 = lda(kr1 + 32, g);

  for (int kv0 = 0; kv0 < kv_end; kv0 += 32) {
    // scores: S[16 x 32] = Q(16x64) @ K^T(64x32)
    v8f s0 = {}, s1 = {};
    s0 = WMMA_BF16(aQ0, bK0, s0);
    s0 = WMMA_BF16(aQ1, bK1, s0);
    s1 = WMMA_BF16(aQ0, bK2, s1);
    s1 = WMMA_BF16(aQ1, bK3, s1);

    // V-tile operands: issue now so they're in flight during softmax VALU work
    const __bf16* vr0 = Vh + (size_t)(l16)      * Ss + kv0;
    const __bf16* vr1 = Vh + (size_t)(16 + l16) * Ss + kv0;
    const __bf16* vr2 = Vh + (size_t)(32 + l16) * Ss + kv0;
    const __bf16* vr3 = Vh + (size_t)(48 + l16) * Ss + kv0;
    v16bf bV0 = lda(vr0, g), bV1 = lda(vr1, g);
    v16bf bV2 = lda(vr2, g), bV3 = lda(vr3, g);

    // next K-tile operands (double buffer; wave-uniform branch)
    const bool has_next = (kv0 + 32) < kv_end;
    v16bf nK0 = bK0, nK1 = bK1, nK2 = bK2, nK3 = bK3;
    if (has_next) {
      const __bf16* nkr0 = Kh + (size_t)(kv0 + 32 + l16) * HD;
      const __bf16* nkr1 = Kh + (size_t)(kv0 + 48 + l16) * HD;
      nK0 = lda(nkr0, g); nK1 = lda(nkr0 + 32, g);
      nK2 = lda(nkr1, g); nK3 = lda(nkr1 + 32, g);
    }

    // causal mask + online softmax; row M = q0 + 8g + r lives across a 16-lane half
#pragma unroll
    for (int r = 0; r < 8; ++r) {
      const int row = q0 + 8 * g + r;
      float x0 = (kv0 + l16 <= row)      ? s0[r] : -__builtin_inff();
      float x1 = (kv0 + 16 + l16 <= row) ? s1[r] : -__builtin_inff();
      float mx = fmaxf(x0, x1);
#pragma unroll
      for (int off = 1; off < 16; off <<= 1) mx = fmaxf(mx, __shfl_xor(mx, off, 32));
      const float mnew = fmaxf(mrow[r], mx);
      const float corr = __expf(mrow[r] - mnew);
      const float p0 = __expf(x0 - mnew);
      const float p1 = __expf(x1 - mnew);
      float sum = p0 + p1;
#pragma unroll
      for (int off = 1; off < 16; off <<= 1) sum += __shfl_xor(sum, off, 32);
      lrow[r] = lrow[r] * corr + sum;
      mrow[r] = mnew;
      o0[r] *= corr; o1[r] *= corr; o2[r] *= corr; o3[r] *= corr;
      plds[wave][8 * g + r][l16]      = (__bf16)p0;
      plds[wave][8 * g + r][16 + l16] = (__bf16)p1;
    }
    // per-wave LDS transpose round-trip (DS ops in-order within a wave)
    asm volatile("s_wait_dscnt 0x0" ::: "memory");
    const __bf16* prow = &plds[wave][l16][0];
    v16bf aP = lda(prow, g);

    // O += P(16x32) @ V(32x64), B gathered from V^T rows (contiguous in kv)
    o0 = WMMA_BF16(aP, bV0, o0);
    o1 = WMMA_BF16(aP, bV1, o1);
    o2 = WMMA_BF16(aP, bV2, o2);
    o3 = WMMA_BF16(aP, bV3, o3);

    bK0 = nK0; bK1 = nK1; bK2 = nK2; bK3 = nK3;
  }

  // normalize + store bf16 attention output row-major [B*S][D]
  const v8f os[4] = {o0, o1, o2, o3};
#pragma unroll
  for (int r = 0; r < 8; ++r) {
    const float inv = 1.0f / lrow[r];
    const int srow  = q0 + 8 * g + r;
    __bf16* dst = Ab + ((size_t)b * Ss + srow) * Dd + h * HD;
#pragma unroll
    for (int j = 0; j < 4; ++j) dst[j * 16 + l16] = (__bf16)(os[j][r] * inv);
  }
}

// ---------------- kernel 5: output projection ----------------
__global__ __launch_bounds__(128) void k_out_gemm(
    const __bf16* __restrict__ Ab, const __bf16* __restrict__ Wt,
    const float* __restrict__ bias, float* __restrict__ out) {
  const int lane = threadIdx.x & 31;
  const int wave = threadIdx.x >> 5;
  const int l16  = lane & 15;
  const int g    = lane >> 4;
  const int m0   = blockIdx.x * 64 + wave * 16;
  const int n0   = blockIdx.y * 64;

  v8f acc0 = {}, acc1 = {}, acc2 = {}, acc3 = {};
  wave_gemm_16x64<Dd>(Ab, Wt, m0, n0, l16, g, acc0, acc1, acc2, acc3);
  const v8f accs[4] = {acc0, acc1, acc2, acc3};

#pragma unroll
  for (int j = 0; j < 4; ++j) {
    const int ng = n0 + j * 16 + l16;
    const float bv = bias[ng];
#pragma unroll
    for (int r = 0; r < 8; ++r) {
      const int mg = m0 + 8 * g + r;
      out[(size_t)mg * Dd + ng] = accs[j][r] + bv;
    }
  }
}

// ---------------- host launch ----------------
extern "C" void kernel_launch(void* const* d_in, const int* in_sizes, int n_in,
                              void* d_out, int out_size, void* d_ws, size_t ws_size,
                              hipStream_t stream) {
  (void)in_sizes; (void)n_in; (void)out_size; (void)ws_size;
  const float* X    = (const float*)d_in[0];
  const float* Wqkv = (const float*)d_in[1];
  const float* bqkv = (const float*)d_in[2];
  const float* Wout = (const float*)d_in[3];
  const float* bout = (const float*)d_in[4];
  float* out = (float*)d_out;

  // workspace layout (56 MiB total)
  char* ws = (char*)d_ws;
  __bf16* Xb   = (__bf16*)(ws);                       // 8 MiB  [4096][1024]
  __bf16* Wqt  = (__bf16*)(ws + (8ull  << 20));       // 6 MiB  [3072][1024]
  __bf16* Wot  = (__bf16*)(ws + (14ull << 20));       // 2 MiB  [1024][1024]
  __bf16* Qb   = (__bf16*)(ws + (16ull << 20));       // 8 MiB  [B*H][S][64]
  __bf16* Kb   = (__bf16*)(ws + (24ull << 20));       // 8 MiB  [B*H][S][64]
  __bf16* Vrow = (__bf16*)(ws + (32ull << 20));       // 8 MiB  [B*H][S][64]
  __bf16* Vt   = (__bf16*)(ws + (40ull << 20));       // 8 MiB  [B*H][64][S]
  __bf16* Ab   = (__bf16*)(ws + (48ull << 20));       // 8 MiB  [4096][1024]

  k_cvt<<<1024, 256, 0, stream>>>((const float4*)X, (v4bf*)Xb, (Mtot * Dd) / 4);

  dim3 gtq(NQKV / 32, Dd / 32);
  k_tr_cvt<<<gtq, 256, 0, stream>>>(Wqkv, Wqt, Dd, NQKV);
  dim3 gto(Dd / 32, Dd / 32);
  k_tr_cvt<<<gto, 256, 0, stream>>>(Wout, Wot, Dd, Dd);

  dim3 gq(Mtot / 64, NQKV / 64);
  k_qkv_gemm<<<gq, 128, 0, stream>>>(Xb, Wqt, bqkv, Qb, Kb, Vrow);

  dim3 gtv(HD / 32, Ss / 32, Bb * Hh);
  k_tr_v<<<gtv, 256, 0, stream>>>(Vrow, Vt);

  k_attn<<<(Bb * Hh * (Ss / 16)) / 4, 128, 0, stream>>>(Qb, Kb, Vt, Ab);

  dim3 go(Mtot / 64, Dd / 64);
  k_out_gemm<<<go, 128, 0, stream>>>(Ab, Wot, bout, out);
}
